// MEMORY_34986803593776
// MI455X (gfx1250) — matrix operand order repaired and checked by reading, not compiled, not run on hardware
//
#include <hip/hip_runtime.h>
#include <hip/hip_bf16.h>
#include <stdint.h>

// ---------------------------------------------------------------------------
// Types for CDNA5 WMMA
// ---------------------------------------------------------------------------
typedef __bf16 bfv16 __attribute__((ext_vector_type(16)));
typedef float  fv8   __attribute__((ext_vector_type(8)));

union FragU { bfv16 v; uint4 q[2]; };

#define WMMA_BF16(a, b, c) \
  __builtin_amdgcn_wmma_f32_16x16x32_bf16(false, (a), false, (b), (short)0, (c), false, false)

__device__ __forceinline__ unsigned short f2bf(float f) {
  unsigned u = __float_as_uint(f);
  u += 0x7FFFu + ((u >> 16) & 1u);
  return (unsigned short)(u >> 16);
}
__device__ __forceinline__ float bf2f(unsigned short h) {
  return __uint_as_float(((unsigned)h) << 16);
}
__device__ __forceinline__ float sigm(float x) { return 1.f / (1.f + __expf(-x)); }

// A-fragment (16x32 bf16, row-major source in LDS).
// Lane L<16: row L, K in [k0..k0+7] U [k0+16..k0+23]
// Lane L>=16: row L-16, K in [k0+8..k0+15] U [k0+24..k0+31]
__device__ __forceinline__ bfv16 loadA_lds(const unsigned short* base, int stride, int lane, int k0) {
  int r = lane & 15;
  int koff = k0 + ((lane & 16) ? 8 : 0);
  FragU f;
  f.q[0] = *reinterpret_cast<const uint4*>(base + r * stride + koff);
  f.q[1] = *reinterpret_cast<const uint4*>(base + r * stride + koff + 16);
  return f.v;
}

// B-fragment (32x16 bf16) from pre-transposed W^T (N x K row-major, global).
// Lane L<16: col N=n0+L, K in [k0..k0+7] U [k0+16..k0+23]; lane L>=16: col n0+L-16, K in [k0+8..k0+15] U [k0+24..k0+31]
__device__ __forceinline__ bfv16 loadB_g(const unsigned short* bt, int strideK, int lane, int n0, int k0) {
  int n = lane & 15;
  int koff = k0 + ((lane & 16) ? 8 : 0);
  const unsigned short* p = bt + (size_t)(n0 + n) * strideK + koff;
  FragU f;
  f.q[0] = *reinterpret_cast<const uint4*>(p);
  f.q[1] = *reinterpret_cast<const uint4*>(p + 16);
  return f.v;
}

// ---------------------------------------------------------------------------
// Weight prep: fp32 -> bf16, transposed (dst is C x R row-major = src^T)
// ---------------------------------------------------------------------------
__global__ void tconv_kernel(const float* __restrict__ src, unsigned short* __restrict__ dst,
                             int R, int C) {
  int i = blockIdx.x * blockDim.x + threadIdx.x;
  if (i < R * C) {
    int r = i / C, c = i % C;
    dst[(size_t)c * R + r] = f2bf(src[i]);
  }
}
__global__ void cconv_kernel(const float* __restrict__ src, unsigned short* __restrict__ dst, int n) {
  int i = blockIdx.x * blockDim.x + threadIdx.x;
  if (i < n) dst[i] = f2bf(src[i]);
}

// ---------------------------------------------------------------------------
// Fused memory-update kernel
// ---------------------------------------------------------------------------
#define TROW 4104  // 4096 + 8 pad (bf16 elems) -> bank-spread for A-frag reads

// LDS layout (bytes)
#define LDS_TILE   0                        // 16 x TROW bf16          = 131328
#define LDS_QA     131328                   // 16 x 128 f32            = 8192
#define LDS_CK     139520                   // 16 x 72 bf16            = 2304
#define LDS_C0     141824                   // 16 x 136 bf16           = 4352
#define LDS_SCORES 146176                   // 16 x 64 f32 (-> weights)= 4096
#define LDS_UE     150272
#define LDS_UZ     154368
#define LDS_UA     158464
#define LDS_VE     162560
#define LDS_VZ     166656
#define LDS_ZT     170752                   // 16 x 72 bf16            = 2304
#define LDS_ER     173056
#define LDS_AD     177152
#define LDS_BYTES  181248

struct Params {
  const float *ck, *qa, *mv;
  const float *be, *bemv, *bza, *bamv, *bc0, *bm1, *bz, *bzmv;
  const unsigned short *mkbf, *wc0t, *wm1t, *w3t, *wezt, *wzat;
  float* out;
};

__global__ __launch_bounds__(256) void memnet_kernel(Params P) {
  extern __shared__ unsigned char smem[];
  unsigned short* tile = (unsigned short*)(smem + LDS_TILE);
  float*          qaT  = (float*)(smem + LDS_QA);
  unsigned short* ckbf = (unsigned short*)(smem + LDS_CK);
  unsigned short* c0bf = (unsigned short*)(smem + LDS_C0);
  float*          wgt  = (float*)(smem + LDS_SCORES);
  float*          Ue   = (float*)(smem + LDS_UE);
  float*          Uz   = (float*)(smem + LDS_UZ);
  float*          Ua   = (float*)(smem + LDS_UA);
  float*          Ve   = (float*)(smem + LDS_VE);
  float*          Vz   = (float*)(smem + LDS_VZ);
  unsigned short* ztbf = (unsigned short*)(smem + LDS_ZT);
  float*          ers  = (float*)(smem + LDS_ER);
  float*          adl  = (float*)(smem + LDS_AD);

  const int tid  = threadIdx.x;
  const int wave = tid >> 5;
  const int lane = tid & 31;
  const int b0   = blockIdx.x * 16;

  // ---- Phase 0: stage memory_value tile (16 x 4096) as bf16, plus qa & ck
  {
    const float* mv = P.mv + (size_t)b0 * 4096;
    for (int j = 0; j < 64; ++j) {
      int pos = tid + j * 256;          // float4 index over 16x4096
      int e = pos * 4;
      int r = e >> 12, c = e & 4095;
      float4 f = *reinterpret_cast<const float4*>(mv + (size_t)r * 4096 + c);
      uint2 pk;
      pk.x = (unsigned)f2bf(f.x) | ((unsigned)f2bf(f.y) << 16);
      pk.y = (unsigned)f2bf(f.z) | ((unsigned)f2bf(f.w) << 16);
      *reinterpret_cast<uint2*>(tile + r * TROW + c) = pk;
    }
    for (int j = 0; j < 2; ++j) {       // qa: 16x128 f32
      int pos = tid + j * 256;
      int r = pos >> 5, c = (pos & 31) * 4;
      float4 f = *reinterpret_cast<const float4*>(P.qa + (size_t)(b0 + r) * 128 + c);
      *reinterpret_cast<float4*>(qaT + r * 128 + c) = f;
    }
    {                                   // ck: 16x64 -> bf16 (A layout, stride 72)
      int r = tid >> 4, c = (tid & 15) * 4;
      float4 f = *reinterpret_cast<const float4*>(P.ck + (size_t)(b0 + r) * 64 + c);
      uint2 pk;
      pk.x = (unsigned)f2bf(f.x) | ((unsigned)f2bf(f.y) << 16);
      pk.y = (unsigned)f2bf(f.z) | ((unsigned)f2bf(f.w) << 16);
      *reinterpret_cast<uint2*>(ckbf + r * 72 + c) = pk;
    }
  }
  __syncthreads();

  // ---- Phase 1: scores = ck @ mk^T  (16x64), waves 0..3
  if (wave < 4) {
    fv8 acc = {0.f, 0.f, 0.f, 0.f, 0.f, 0.f, 0.f, 0.f};
    for (int kk = 0; kk < 2; ++kk) {
      bfv16 a = loadA_lds(ckbf, 72, lane, kk * 32);
      bfv16 b = loadB_g(P.mkbf, 64, lane, wave * 16, kk * 32);
      acc = WMMA_BF16(a, b, acc);
    }
    int n = wave * 16 + (lane & 15);
    int rb = (lane & 16) ? 8 : 0;
    for (int j = 0; j < 8; ++j) wgt[(rb + j) * 64 + n] = acc[j];
  }
  __syncthreads();

  // ---- Phase 2: row softmax over M=64 (threads 0..15, one row each)
  if (tid < 16) {
    float* s = wgt + tid * 64;
    float mx = -1e30f;
    for (int c = 0; c < 64; ++c) mx = fmaxf(mx, s[c]);
    float sum = 0.f;
    for (int c = 0; c < 64; ++c) { float e = __expf(s[c] - mx); s[c] = e; sum += e; }
    float inv = 1.f / sum;
    for (int c = 0; c < 64; ++c) s[c] *= inv;
  }
  __syncthreads();

  // ---- Phase 3: content0 = qa * sigmoid(mem_flat0 @ Wc0 + bc0)   (K=4096, N=128)
  {
    fv8 acc = {0.f, 0.f, 0.f, 0.f, 0.f, 0.f, 0.f, 0.f};
    int n0 = wave * 16;
#pragma unroll 2
    for (int k = 0; k < 128; ++k) {
      bfv16 a = loadA_lds(tile, TROW, lane, k * 32);
      bfv16 b = loadB_g(P.wc0t, 4096, lane, n0, k * 32);
      acc = WMMA_BF16(a, b, acc);
    }
    int n = n0 + (lane & 15);
    int rb = (lane & 16) ? 8 : 0;
    float bcv = P.bc0[n];
    for (int j = 0; j < 8; ++j) {
      int r = rb + j;
      float c0 = qaT[r * 128 + n] * sigm(acc[j] + bcv);
      c0bf[r * 136 + n] = f2bf(c0);
    }
  }
  __syncthreads();

  // ---- Phase 4: gate1 = sigmoid(content0 @ Wm1 + bm1); tile *= gate1 (in place)
  {
    bfv16 a0 = loadA_lds(c0bf, 136, lane, 0);
    bfv16 a1 = loadA_lds(c0bf, 136, lane, 32);
    bfv16 a2 = loadA_lds(c0bf, 136, lane, 64);
    bfv16 a3 = loadA_lds(c0bf, 136, lane, 96);
    for (int t = 0; t < 32; ++t) {
      int n0 = (wave * 32 + t) * 16;
      fv8 acc = {0.f, 0.f, 0.f, 0.f, 0.f, 0.f, 0.f, 0.f};
      acc = WMMA_BF16(a0, loadB_g(P.wm1t, 128, lane, n0, 0),  acc);
      acc = WMMA_BF16(a1, loadB_g(P.wm1t, 128, lane, n0, 32), acc);
      acc = WMMA_BF16(a2, loadB_g(P.wm1t, 128, lane, n0, 64), acc);
      acc = WMMA_BF16(a3, loadB_g(P.wm1t, 128, lane, n0, 96), acc);
      int n = n0 + (lane & 15);
      int rb = (lane & 16) ? 8 : 0;
      float bm = P.bm1[n];
      for (int j = 0; j < 8; ++j) {
        int r = rb + j;
        unsigned short* p = tile + r * TROW + n;
        float g = sigm(acc[j] + bm);
        *p = f2bf(bf2f(*p) * g);
      }
    }
  }
  __syncthreads();

  // ---- Phase 5: U{e,z,a} = mem_flat @ W{e,z,a}mv  (K=4096, N=3x64 concat)
  for (int nt = wave; nt < 12; nt += 8) {
    fv8 acc = {0.f, 0.f, 0.f, 0.f, 0.f, 0.f, 0.f, 0.f};
#pragma unroll 2
    for (int k = 0; k < 128; ++k) {
      bfv16 a = loadA_lds(tile, TROW, lane, k * 32);
      bfv16 b = loadB_g(P.w3t, 4096, lane, nt * 16, k * 32);
      acc = WMMA_BF16(a, b, acc);
    }
    float* dst = (nt < 4) ? Ue : ((nt < 8) ? Uz : Ua);
    int n = (nt & 3) * 16 + (lane & 15);
    int rb = (lane & 16) ? 8 : 0;
    for (int j = 0; j < 8; ++j) dst[(rb + j) * 64 + n] = acc[j];
  }
  // ---- Phase 5b: V{e,z} = content0 @ W{e,z}  (K=128, N=2x64 concat), wave==nt
  {
    int nt = wave;  // 0..7
    fv8 acc = {0.f, 0.f, 0.f, 0.f, 0.f, 0.f, 0.f, 0.f};
    for (int kk = 0; kk < 4; ++kk) {
      bfv16 a = loadA_lds(c0bf, 136, lane, kk * 32);
      bfv16 b = loadB_g(P.wezt, 128, lane, nt * 16, kk * 32);
      acc = WMMA_BF16(a, b, acc);
    }
    float* dst = (nt < 4) ? Ve : Vz;
    int n = (nt & 3) * 16 + (lane & 15);
    int rb = (lane & 16) ? 8 : 0;
    for (int j = 0; j < 8; ++j) dst[(rb + j) * 64 + n] = acc[j];
  }
  __syncthreads();

  // ---- Phase 6: zt (bf16) and erase_signal (f32), elementwise 16x64
  for (int i = tid; i < 1024; i += 256) {
    int r = i >> 6, v = i & 63;
    float z = sigm(Vz[i] + P.bz[v] + Uz[i] + P.bzmv[v]);
    ztbf[r * 72 + v] = f2bf(z);
    ers[i] = sigm(sigm(Ve[i] + P.be[v]) + sigm(Ue[i] + P.bemv[v]));
  }
  __syncthreads();

  // ---- Phase 7: add_signal = tanh(tanh(zt @ Wza + bza) + tanh(Ua + bamv))
  if (wave < 4) {
    fv8 acc = {0.f, 0.f, 0.f, 0.f, 0.f, 0.f, 0.f, 0.f};
    for (int kk = 0; kk < 2; ++kk) {
      bfv16 a = loadA_lds(ztbf, 72, lane, kk * 32);
      bfv16 b = loadB_g(P.wzat, 64, lane, wave * 16, kk * 32);
      acc = WMMA_BF16(a, b, acc);
    }
    int n = wave * 16 + (lane & 15);
    int rb = (lane & 16) ? 8 : 0;
    float bzav = P.bza[n], bamvv = P.bamv[n];
    for (int j = 0; j < 8; ++j) {
      int r = rb + j, i = r * 64 + n;
      adl[i] = tanhf(tanhf(acc[j] + bzav) + tanhf(Ua[i] + bamvv));
    }
  }
  __syncthreads();

  // ---- Phase 8: new_memory = pre*(1 - w*erase) + w*add  (fused store)
  {
    float* outp = P.out + (size_t)b0 * 4096;
    for (int j = 0; j < 64; ++j) {
      int pos = tid + j * 256;
      int e = pos * 4;
      int r = e >> 12, c = e & 4095;
      int m = c >> 6, v = c & 63;
      float w = wgt[r * 64 + m];
      uint2 pk = *reinterpret_cast<const uint2*>(tile + r * TROW + c);
      float p0 = bf2f((unsigned short)(pk.x & 0xFFFF));
      float p1 = bf2f((unsigned short)(pk.x >> 16));
      float p2 = bf2f((unsigned short)(pk.y & 0xFFFF));
      float p3 = bf2f((unsigned short)(pk.y >> 16));
      const float* er = ers + r * 64 + v;
      const float* ad = adl + r * 64 + v;
      float4 o;
      o.x = p0 * (1.f - w * er[0]) + w * ad[0];
      o.y = p1 * (1.f - w * er[1]) + w * ad[1];
      o.z = p2 * (1.f - w * er[2]) + w * ad[2];
      o.w = p3 * (1.f - w * er[3]) + w * ad[3];
      *reinterpret_cast<float4*>(outp + (size_t)r * 4096 + c) = o;
    }
  }
}

// ---------------------------------------------------------------------------
// Launch
// ---------------------------------------------------------------------------
// workspace layout (bf16 element offsets)
#define WS_MK   0            // 64 x 64
#define WS_WC0T 4096         // 128 x 4096
#define WS_WM1T 528384       // 4096 x 128
#define WS_W3T  1052672      // 192 x 4096  (Wemv^T ; Wzmv^T ; Wamv^T)
#define WS_WEZT 1839104      // 128 x 128   (We^T ; Wz^T)
#define WS_WZAT 1855488      // 64 x 64

extern "C" void kernel_launch(void* const* d_in, const int* in_sizes, int n_in,
                              void* d_out, int out_size, void* d_ws, size_t ws_size,
                              hipStream_t stream) {
  const float* ck   = (const float*)d_in[0];
  const float* qa   = (const float*)d_in[1];
  const float* mk   = (const float*)d_in[2];
  const float* mv   = (const float*)d_in[3];
  const float* We   = (const float*)d_in[4];
  const float* be   = (const float*)d_in[5];
  const float* Wemv = (const float*)d_in[6];
  const float* bemv = (const float*)d_in[7];
  const float* Wza  = (const float*)d_in[8];
  const float* bza  = (const float*)d_in[9];
  const float* Wamv = (const float*)d_in[10];
  const float* bamv = (const float*)d_in[11];
  const float* Wc0  = (const float*)d_in[12];
  const float* bc0  = (const float*)d_in[13];
  const float* Wm1  = (const float*)d_in[14];
  const float* bm1  = (const float*)d_in[15];
  const float* Wz   = (const float*)d_in[16];
  const float* bz   = (const float*)d_in[17];
  const float* Wzmv = (const float*)d_in[18];
  const float* bzmv = (const float*)d_in[19];

  unsigned short* ws = (unsigned short*)d_ws;

  // Weight prep: bf16 transposed copies (deterministic, recomputed each call)
  cconv_kernel<<<16, 256, 0, stream>>>(mk, ws + WS_MK, 64 * 64);
  tconv_kernel<<<2048, 256, 0, stream>>>(Wc0, ws + WS_WC0T, 4096, 128);
  tconv_kernel<<<2048, 256, 0, stream>>>(Wm1, ws + WS_WM1T, 128, 4096);
  tconv_kernel<<<1024, 256, 0, stream>>>(Wemv, ws + WS_W3T, 4096, 64);
  tconv_kernel<<<1024, 256, 0, stream>>>(Wzmv, ws + WS_W3T + 64 * 4096, 4096, 64);
  tconv_kernel<<<1024, 256, 0, stream>>>(Wamv, ws + WS_W3T + 128 * 4096, 4096, 64);
  tconv_kernel<<<32, 256, 0, stream>>>(We, ws + WS_WEZT, 128, 64);
  tconv_kernel<<<32, 256, 0, stream>>>(Wz, ws + WS_WEZT + 64 * 128, 128, 64);
  tconv_kernel<<<16, 256, 0, stream>>>(Wza, ws + WS_WZAT, 64, 64);

  Params P;
  P.ck = ck; P.qa = qa; P.mv = mv;
  P.be = be; P.bemv = bemv; P.bza = bza; P.bamv = bamv;
  P.bc0 = bc0; P.bm1 = bm1; P.bz = bz; P.bzmv = bzmv;
  P.mkbf = ws + WS_MK; P.wc0t = ws + WS_WC0T; P.wm1t = ws + WS_WM1T;
  P.w3t = ws + WS_W3T; P.wezt = ws + WS_WEZT; P.wzat = ws + WS_WZAT;
  P.out = (float*)d_out;

  int B = in_sizes[0] / 64;  // control_key is (B, 64)
  int grid = B / 16;
  memnet_kernel<<<grid, 256, LDS_BYTES, stream>>>(P);

  (void)n_in; (void)out_size; (void)ws_size;
}
